// FeaturesSelfAttention_35141422416316
// MI455X (gfx1250) — compile-verified, matrix operand order checked by reading, not run on hardware
//
#include <hip/hip_runtime.h>
#include <hip/hip_bf16.h>

typedef __attribute__((ext_vector_type(16))) _Float16 v16h;
typedef __attribute__((ext_vector_type(8)))  _Float16 v8h;
typedef __attribute__((ext_vector_type(8)))  float    v8f;
typedef __attribute__((ext_vector_type(4)))  float    f32x4;
typedef __attribute__((ext_vector_type(4)))  unsigned int u32x4;

union FragAB { v16h f; u32x4 u[2]; };
union Frag8  { v8h  h; u32x4 u;    };

#define NFEAT 13
#define HID   128
#define NHEAD 4
#define HDIM  32

// ---------------------------------------------------------------------------
// Prep: convert Wq|Wk|Wv (3 x 128x128 f32, row-major) to f16 in workspace.
// Row-major f16 is already the correct source layout for WMMA B fragments:
// lane n reads W[n][kc*32 + g*8 .. +7] and W[n][kc*32 + 16 + g*8 .. +7].
// ---------------------------------------------------------------------------
__global__ void convert_w_kernel(const float* __restrict__ Wq,
                                 const float* __restrict__ Wk,
                                 const float* __restrict__ Wv,
                                 _Float16* __restrict__ dst) {
  int i = blockIdx.x * blockDim.x + threadIdx.x;
  if (i >= 3 * HID * HID) return;
  const float* src = (i < HID * HID) ? Wq : (i < 2 * HID * HID) ? Wk : Wv;
  dst[i] = (_Float16)src[i & (HID * HID - 1)];
}

// ---------------------------------------------------------------------------
// Main: one wave32 handles TWO batch items (B-fragments loaded once, used by
// two independent WMMA chains). 8 waves per 256-thread block.
// LDS per wave per item (halves): q 16x32 (512) | k 16x32 (512)
//                                 | v^T 32x16 (512) | probs 16x16 (256) = 1792
// ---------------------------------------------------------------------------
__global__ __launch_bounds__(256) void attn_kernel(
    const int*   __restrict__ hs,
    const float* __restrict__ num_emb,
    const float* __restrict__ feat_emb,
    const _Float16* __restrict__ wf,
    const float* __restrict__ bq,
    const float* __restrict__ bk,
    const float* __restrict__ bv,
    float* __restrict__ out, int batch, int vocab) {
  __shared__ __align__(16) _Float16 smem[8 * 2 * 1792];

  const int lane = threadIdx.x & 31;
  const int wave = threadIdx.x >> 5;
  const int widx = blockIdx.x * 8 + wave;
  const int i0   = widx * 2;
  if (i0 >= batch) return;                    // wave-uniform exit
  const bool has2 = (i0 + 1 < batch);
  const int  i1   = has2 ? i0 + 1 : i0;

  _Float16* base = smem + wave * (2 * 1792);
  _Float16* qbuf[2]  = { base,        base + 1792 };
  _Float16* kbuf[2]  = { base + 512,  base + 1792 + 512 };
  _Float16* vt[2]    = { base + 1024, base + 1792 + 1024 };   // transposed [d][s]
  _Float16* probs[2] = { base + 1536, base + 1792 + 1536 };

  const int m = lane & 15;   // A row / B column within a tile
  const int g = lane >> 4;   // lane group (selects K sub-block per ISA layout)
  u32x4 zero4 = {};

  // ---- build emb A-fragments directly from global (no LDS round trip).
  // lane (m,g) supplies row m; rows >= 13 are zeroed via fsc = 0.
  FragAB A[2][4];
  {
    const float fsc  = (m < NFEAT) ? 0.5f : 0.0f;
    const int   mrow = (m < NFEAT) ? m : 0;
    const float* frow = feat_emb + mrow * HID;
    const int items[2] = { i0, i1 };
#pragma unroll
    for (int e = 0; e < 2; ++e) {
      int raw = hs[(size_t)items[e] * NFEAT + mrow];
      int ix  = (raw > -1) ? raw : (vocab - 1);
      const float* nrow = num_emb + (size_t)((m < NFEAT) ? ix : 0) * HID;
#pragma unroll
      for (int kc = 0; kc < 4; ++kc)
#pragma unroll
        for (int blk = 0; blk < 2; ++blk) {
          const int off = kc * 32 + blk * 16 + g * 8;
          f32x4 a0 = *(const f32x4*)(nrow + off);
          f32x4 a1 = *(const f32x4*)(nrow + off + 4);
          f32x4 f0 = *(const f32x4*)(frow + off);
          f32x4 f1 = *(const f32x4*)(frow + off + 4);
          f32x4 e0 = (a0 + f0) * fsc;
          f32x4 e1 = (a1 + f1) * fsc;
          Frag8 pk;
          pk.h[0] = (_Float16)e0.x; pk.h[1] = (_Float16)e0.y;
          pk.h[2] = (_Float16)e0.z; pk.h[3] = (_Float16)e0.w;
          pk.h[4] = (_Float16)e1.x; pk.h[5] = (_Float16)e1.y;
          pk.h[6] = (_Float16)e1.z; pk.h[7] = (_Float16)e1.w;
          A[e][kc].u[blk] = pk.u;
        }
    }
  }

  for (int h = 0; h < NHEAD; ++h) {
    // ---- q/k/v projections: out[s][n] = sum_k emb[s][k]*W[n][k] + b[n]
    // B fragment loaded once, consumed by both items' WMMA chains.
#pragma unroll
    for (int p = 0; p < 3; ++p) {
      const float* bias = (p == 0) ? bq : (p == 1) ? bk : bv;
#pragma unroll
      for (int t = 0; t < 2; ++t) {
        const int n = h * HDIM + t * 16 + m;                  // output column
        const _Float16* wrow = wf + ((size_t)p << 14) + (size_t)n * HID;
        v8f acc0 = {}, acc1 = {};
#pragma unroll
        for (int kc = 0; kc < 4; ++kc) {
          FragAB B;
          B.u[0] = *(const u32x4*)(wrow + kc * 32 + g * 8);
          B.u[1] = *(const u32x4*)(wrow + kc * 32 + 16 + g * 8);
          acc0 = __builtin_amdgcn_wmma_f32_16x16x32_f16(false, A[0][kc].f, false, B.f,
                                                        (short)0, acc0, false, false);
          acc1 = __builtin_amdgcn_wmma_f32_16x16x32_f16(false, A[1][kc].f, false, B.f,
                                                        (short)0, acc1, false, false);
        }
        const float bn = bias[n];
        const int d = t * 16 + m;                             // column within head
#pragma unroll
        for (int e = 0; e < 2; ++e) {
          const v8f& acc = e ? acc1 : acc0;
          if (p < 2) {                                        // q,k: row-major [s][d]
            _Float16* dst = (p == 0) ? qbuf[e] : kbuf[e];
#pragma unroll
            for (int r = 0; r < 8; ++r)
              dst[(r + 8 * g) * HDIM + d] = (_Float16)(acc[r] + bn);
          } else {                                            // v: transposed [d][s]
            Frag8 pv;
#pragma unroll
            for (int r = 0; r < 8; ++r) pv.h[r] = (_Float16)(acc[r] + bn);
            *(u32x4*)(vt[e] + d * 16 + 8 * g) = pv.u;         // one b128 store
          }
        }
      }
    }
    __builtin_amdgcn_fence(__ATOMIC_ACQ_REL, "wavefront");

    // ---- scores = q @ k^T (K = 32), then softmax over keys (mask n >= 13)
#pragma unroll
    for (int e = 0; e < 2; ++e) {
      FragAB qa, kb;
      const _Float16* qrow = qbuf[e] + m * HDIM;
      const _Float16* krow = kbuf[e] + m * HDIM;
      qa.u[0] = *(const u32x4*)(qrow + g * 8);
      qa.u[1] = *(const u32x4*)(qrow + 16 + g * 8);
      kb.u[0] = *(const u32x4*)(krow + g * 8);
      kb.u[1] = *(const u32x4*)(krow + 16 + g * 8);
      v8f sc = {};
      sc = __builtin_amdgcn_wmma_f32_16x16x32_f16(false, qa.f, false, kb.f,
                                                  (short)0, sc, false, false);
      const float scale = 0.17677669529663687f;  // 1/sqrt(32)
      float x[8], mx[8], pr[8], sum[8];
#pragma unroll
      for (int r = 0; r < 8; ++r) {
        x[r]  = (m < NFEAT) ? sc[r] * scale : -1.0e30f;
        mx[r] = x[r];
      }
#pragma unroll
      for (int off = 1; off < 16; off <<= 1)
#pragma unroll
        for (int r = 0; r < 8; ++r)
          mx[r] = fmaxf(mx[r], __shfl_xor(mx[r], off, 32));
#pragma unroll
      for (int r = 0; r < 8; ++r) { pr[r] = __expf(x[r] - mx[r]); sum[r] = pr[r]; }
#pragma unroll
      for (int off = 1; off < 16; off <<= 1)
#pragma unroll
        for (int r = 0; r < 8; ++r)
          sum[r] += __shfl_xor(sum[r], off, 32);
#pragma unroll
      for (int r = 0; r < 8; ++r)
        probs[e][(r + 8 * g) * 16 + m] = (_Float16)(pr[r] / sum[r]);
    }
    __builtin_amdgcn_fence(__ATOMIC_ACQ_REL, "wavefront");

    // ---- ctx = probs @ v, then mean over s=0..12, write out
#pragma unroll
    for (int e = 0; e < 2; ++e) {
      FragAB pa;
      pa.u[0] = *(const u32x4*)(probs[e] + m * 16 + g * 8);
      pa.u[1] = zero4;                      // K = 16..31 zero-padded
#pragma unroll
      for (int t = 0; t < 2; ++t) {
        FragAB vb;
        vb.u[0] = *(const u32x4*)(vt[e] + (t * 16 + m) * 16 + g * 8);
        vb.u[1] = zero4;                    // avoid 0 x NaN
        v8f cx = {};
        cx = __builtin_amdgcn_wmma_f32_16x16x32_f16(false, pa.f, false, vb.f,
                                                    (short)0, cx, false, false);
        // rows held by this lane: m_row = r + 8*g; valid rows are < 13
        float part = cx[0] + cx[1] + cx[2] + cx[3] + cx[4];
        if (g == 0) part += cx[5] + cx[6] + cx[7];
        part += __shfl_xor(part, 16, 32);
        if (lane < 16 && (e == 0 || has2))
          out[(size_t)(e ? i1 : i0) * HID + h * HDIM + t * 16 + lane] =
              part * (1.0f / 13.0f);
      }
    }
    __builtin_amdgcn_fence(__ATOMIC_ACQ_REL, "wavefront");  // buffers reused next head
  }
}

// ---------------------------------------------------------------------------
extern "C" void kernel_launch(void* const* d_in, const int* in_sizes, int n_in,
                              void* d_out, int out_size, void* d_ws, size_t ws_size,
                              hipStream_t stream) {
  const int*   hs       = (const int*)d_in[0];
  const float* num_emb  = (const float*)d_in[1];
  const float* feat_emb = (const float*)d_in[2];
  const float* Wq       = (const float*)d_in[3];
  const float* bq       = (const float*)d_in[4];
  const float* Wk       = (const float*)d_in[5];
  const float* bk       = (const float*)d_in[6];
  const float* Wv       = (const float*)d_in[7];
  const float* bv       = (const float*)d_in[8];
  float*       out      = (float*)d_out;

  const int batch = in_sizes[0] / NFEAT;
  const int vocab = in_sizes[1] / HID;

  _Float16* wf = (_Float16*)d_ws;          // 3*128*128*2 = 96 KB of workspace

  const int nW = 3 * HID * HID;
  hipLaunchKernelGGL(convert_w_kernel, dim3((nW + 255) / 256), dim3(256), 0, stream,
                     Wq, Wk, Wv, wf);

  // 8 waves per block, 2 items per wave -> 16 items per block
  const int blocks = (batch + 15) / 16;
  hipLaunchKernelGGL(attn_kernel, dim3(blocks), dim3(256), 0, stream,
                     hs, num_emb, feat_emb, wf, bq, bk, bv, out, batch, vocab);
}
